// AttentionPoolingClassifier_69741678952669
// MI455X (gfx1250) — compile-verified, matrix operand order checked by reading, not run on hardware
//
#include <hip/hip_runtime.h>
#include <hip/hip_bf16.h>

#define BB   32
#define NN   4096
#define CC   768
#define HH   12
#define HP   16          // heads padded to 16 for WMMA tiles
#define HD   64
#define OUTD 1000
#define EPSN 1e-6f

#define STATS_BLOCKS 512
#define ROWS_PER_BLK ((BB * NN) / STATS_BLOCKS)   // 256

typedef __attribute__((ext_vector_type(16))) _Float16 v16h;
typedef __attribute__((ext_vector_type(8)))  float    v8f;

// ---------------- workspace layout (float offsets) ----------------
#define OFF_PSUM   0L
#define OFF_PSUMSQ (OFF_PSUM   + (long)STATS_BLOCKS * CC)
#define OFF_MEAN   (OFF_PSUMSQ + (long)STATS_BLOCKS * CC)
#define OFF_RSTD   (OFF_MEAN + CC)
#define OFF_BIAS   (OFF_RSTD + CC)
#define OFF_W2H    (OFF_BIAS + 16)                    // 768*16 halves = 6144 floats
#define OFF_W2F    (OFF_W2H + (CC * HP) / 2)          // swizzled frags: 24*32*16 halves
#define OFF_ATTN   (OFF_W2F + (24 * 32 * 16) / 2)
#define OFF_POOL   (OFF_ATTN + (long)BB * HP * NN)
// end = OFF_POOL + BB*HH*CC  (~13 MB total)

// ---------------- 1) per-channel partial sums (deterministic) ---------------
__global__ void k_bn_partial(const float* __restrict__ x,
                             float* __restrict__ psum, float* __restrict__ psumsq) {
  int tid = threadIdx.x;            // 0..255
  int blk = blockIdx.x;             // 0..511
  const float* base = x + (long)blk * ROWS_PER_BLK * CC;
  int c0 = tid, c1 = tid + 256, c2 = tid + 512;
  float s0 = 0.f, s1 = 0.f, s2 = 0.f, q0 = 0.f, q1 = 0.f, q2 = 0.f;
  for (int r = 0; r < ROWS_PER_BLK; ++r) {
    const float* p = base + (long)r * CC;
    float a = p[c0], b = p[c1], c = p[c2];
    s0 += a; q0 += a * a;
    s1 += b; q1 += b * b;
    s2 += c; q2 += c * c;
  }
  long o = (long)blk * CC;
  psum[o + c0] = s0; psumsq[o + c0] = q0;
  psum[o + c1] = s1; psumsq[o + c1] = q1;
  psum[o + c2] = s2; psumsq[o + c2] = q2;
}

// ---------------- 2) finalize stats + fold BN and q into Wk -> W2 (f16) -----
__global__ void k_finalize(const float* __restrict__ Wk, const float* __restrict__ cls,
                           const float* __restrict__ psum, const float* __restrict__ psumsq,
                           float* __restrict__ mean, float* __restrict__ rstd,
                           _Float16* __restrict__ W2h) {
  int c = blockIdx.x * blockDim.x + threadIdx.x;   // 0..767
  if (c >= CC) return;
  float s = 0.f, q = 0.f;
  for (int b = 0; b < STATS_BLOCKS; ++b) {
    s += psum[(long)b * CC + c];
    q += psumsq[(long)b * CC + c];
  }
  const float invn = 1.0f / (float)(BB * NN);
  float m   = s * invn;
  float var = q * invn - m * m;
  float rs  = rsqrtf(var + EPSN);
  mean[c] = m; rstd[c] = rs;
  const float scale = 0.125f;   // hd^-0.5 = 64^-0.5
  for (int h = 0; h < HP; ++h) {
    float w = 0.f;
    if (h < HH) {
      const float* wr = Wk + (long)c * CC + h * HD;
      const float* cr = cls + h * HD;
      for (int d = 0; d < HD; ++d) w += wr[d] * cr[d];
      w *= rs * scale;
    }
    W2h[c * HP + h] = (_Float16)w;   // zero-padded cols 12..15
  }
}

// ---------------- 3) bias[h] = mean . W2[:,h]  +  swizzle W2 into B frags ---
__global__ void k_bias(const float* __restrict__ mean, const _Float16* __restrict__ W2h,
                       float* __restrict__ bias) {
  int h = threadIdx.x;
  if (h >= HP) return;
  float s = 0.f;
  for (int c = 0; c < CC; ++c) s += mean[c] * (float)W2h[c * HP + h];
  bias[h] = s;
}

// W2frag[(tile*32 + lane)*16 + e] holds the e-th half of lane's B fragment for
// K-tile `tile` (ISA 16-bit B layout: n = lane&15, K split by half-wave).
__global__ void k_swizzle_w2(const _Float16* __restrict__ W2h,
                             _Float16* __restrict__ W2f) {
  int idx = blockIdx.x * blockDim.x + threadIdx.x;   // 0..767 (24 tiles * 32 lanes)
  if (idx >= 24 * 32) return;
  int tile = idx >> 5;
  int lane = idx & 31;
  int n  = lane & 15;
  int lo = (lane >> 4) * 8;
#pragma unroll
  for (int e = 0; e < 16; ++e) {
    int kl = (e < 8) ? (lo + e) : (16 + lo + (e - 8));
    W2f[(long)idx * 16 + e] = W2h[(tile * 32 + kl) * HP + n];
  }
}

// ---------------- 4) scores = x @ W2 - bias, via WMMA f16 -------------------
// one wave -> one 16(row)x16(head) tile, K loop over 768 (24 steps)
__global__ void k_scores_wmma(const float* __restrict__ x, const _Float16* __restrict__ W2f,
                              const float* __restrict__ bias, float* __restrict__ attn) {
  int wave = (int)((blockIdx.x * blockDim.x + threadIdx.x) >> 5);  // 0..8191
  int lane = threadIdx.x & 31;
  int m  = lane & 15;
  int hi = lane >> 4;
  int lo = hi * 8;
  long rowbase = (long)wave * 16;
  const float* arow = x + (rowbase + m) * CC;
  v8f acc = {};
  for (int k0 = 0; k0 < CC; k0 += 32) {
    const float4* pa0 = (const float4*)(arow + k0 + lo);
    const float4* pa1 = (const float4*)(arow + k0 + 16 + lo);
    float4 f0 = pa0[0], f1 = pa0[1], f2 = pa1[0], f3 = pa1[1];
    v16h a;
    a[0]=(_Float16)f0.x;  a[1]=(_Float16)f0.y;  a[2]=(_Float16)f0.z;  a[3]=(_Float16)f0.w;
    a[4]=(_Float16)f1.x;  a[5]=(_Float16)f1.y;  a[6]=(_Float16)f1.z;  a[7]=(_Float16)f1.w;
    a[8]=(_Float16)f2.x;  a[9]=(_Float16)f2.y;  a[10]=(_Float16)f2.z; a[11]=(_Float16)f2.w;
    a[12]=(_Float16)f3.x; a[13]=(_Float16)f3.y; a[14]=(_Float16)f3.z; a[15]=(_Float16)f3.w;
    v16h bm = *(const v16h*)(W2f + ((long)((k0 >> 5) * 32 + lane)) * 16);
    acc = __builtin_amdgcn_wmma_f32_16x16x32_f16(false, a, false, bm, (short)0, acc,
                                                 false, false);
  }
  // per-lane output: column n = lane&15, 8 consecutive rows starting at 8*hi
  int n = m;
  float bn = bias[n];
  long rr0 = rowbase + 8 * hi;             // multiple of 8, no batch straddle
  int b  = (int)(rr0 >> 12);               // / 4096
  int nn = (int)(rr0 & 4095);
  float* dst = attn + ((long)(b * HP + n)) * NN + nn;
  float4 s0, s1;
  s0.x = acc[0] - bn; s0.y = acc[1] - bn; s0.z = acc[2] - bn; s0.w = acc[3] - bn;
  s1.x = acc[4] - bn; s1.y = acc[5] - bn; s1.z = acc[6] - bn; s1.w = acc[7] - bn;
  *(float4*)dst       = s0;
  *((float4*)dst + 1) = s1;
}

// ---------------- 5) softmax over N per (b,h_pad), in place -----------------
__global__ void k_softmax(float* __restrict__ attn) {
  __shared__ float red[256];
  float* p = attn + (long)blockIdx.x * NN;
  int t = threadIdx.x;
  float mx = -3.0e38f;
  for (int i = t; i < NN; i += 256) mx = fmaxf(mx, p[i]);
  red[t] = mx; __syncthreads();
  for (int s = 128; s > 0; s >>= 1) { if (t < s) red[t] = fmaxf(red[t], red[t + s]); __syncthreads(); }
  mx = red[0]; __syncthreads();
  float sum = 0.f;
  for (int i = t; i < NN; i += 256) { float e = __expf(p[i] - mx); p[i] = e; sum += e; }
  red[t] = sum; __syncthreads();
  for (int s = 128; s > 0; s >>= 1) { if (t < s) red[t] += red[t + s]; __syncthreads(); }
  float inv = 1.0f / red[0];
  for (int i = t; i < NN; i += 256) p[i] *= inv;
}

// ---------------- 6) pooled = (attn @ x - mean) * rstd, via WMMA ------------
// one wave -> one 16(head)x16(channel) tile per batch, K loop over 4096
__global__ void k_pool_wmma(const float* __restrict__ x, const float* __restrict__ attn,
                            const float* __restrict__ mean, const float* __restrict__ rstd,
                            float* __restrict__ pooled) {
  int wave = (int)((blockIdx.x * blockDim.x + threadIdx.x) >> 5);  // 0..1535
  int lane = threadIdx.x & 31;
  int b  = wave / (CC / 16);
  int ct = wave % (CC / 16);
  int c0 = ct * 16;
  int m  = lane & 15;
  int hi = lane >> 4;
  int lo = hi * 8;
  const float* arow = attn + ((long)(b * HP + m)) * NN;   // padded heads: no guard
  const float* xb   = x + (long)b * NN * CC;
  v8f acc = {};
  for (int k0 = 0; k0 < NN; k0 += 32) {
    const float4* pa0 = (const float4*)(arow + k0 + lo);
    const float4* pa1 = (const float4*)(arow + k0 + 16 + lo);
    float4 f0 = pa0[0], f1 = pa0[1], f2 = pa1[0], f3 = pa1[1];
    v16h a;
    a[0]=(_Float16)f0.x;  a[1]=(_Float16)f0.y;  a[2]=(_Float16)f0.z;  a[3]=(_Float16)f0.w;
    a[4]=(_Float16)f1.x;  a[5]=(_Float16)f1.y;  a[6]=(_Float16)f1.z;  a[7]=(_Float16)f1.w;
    a[8]=(_Float16)f2.x;  a[9]=(_Float16)f2.y;  a[10]=(_Float16)f2.z; a[11]=(_Float16)f2.w;
    a[12]=(_Float16)f3.x; a[13]=(_Float16)f3.y; a[14]=(_Float16)f3.z; a[15]=(_Float16)f3.w;
    v16h bm;
#pragma unroll
    for (int j = 0; j < 8; ++j) {
      int kk = ((j < 4) ? (2 * j) : (16 + 2 * (j - 4))) + lo;
      bm[2 * j]     = (_Float16)xb[(long)(k0 + kk) * CC + c0 + m];
      bm[2 * j + 1] = (_Float16)xb[(long)(k0 + kk + 1) * CC + c0 + m];
    }
    acc = __builtin_amdgcn_wmma_f32_16x16x32_f16(false, a, false, bm, (short)0, acc,
                                                 false, false);
  }
  int c = c0 + m;
  float mc = mean[c], rc = rstd[c];
#pragma unroll
  for (int r = 0; r < 8; ++r) {
    int h = r + 8 * hi;
    if (h < HH) pooled[((long)(b * HH + h)) * CC + c] = (acc[r] - mc) * rc;
  }
}

// ---------------- 7) x_cls[b,c] = pooled[b,h(c),:] . Wv[:,c] ----------------
__global__ void k_xcls(const float* __restrict__ pooled, const float* __restrict__ Wv,
                       float* __restrict__ xcls_out) {
  int idx = blockIdx.x * blockDim.x + threadIdx.x;
  if (idx >= BB * CC) return;
  int b = idx / CC, c = idx % CC, h = c / HD;
  const float* pr = pooled + ((long)(b * HH + h)) * CC;
  float acc = 0.f;
  for (int k = 0; k < CC; ++k) acc += pr[k] * Wv[(long)k * CC + c];
  xcls_out[(long)b * CC + c] = acc;
}

// ---------------- 8) out = x_cls @ Wlin + blin ------------------------------
__global__ void k_out(const float* __restrict__ xcls, const float* __restrict__ Wlin,
                      const float* __restrict__ blin, float* __restrict__ out) {
  int idx = blockIdx.x * blockDim.x + threadIdx.x;
  if (idx >= BB * OUTD) return;
  int b = idx / OUTD, o = idx % OUTD;
  const float* xr = xcls + (long)b * CC;
  float acc = blin[o];
  for (int k = 0; k < CC; ++k) acc += xr[k] * Wlin[(long)k * OUTD + o];
  out[(long)b * OUTD + o] = acc;
}

extern "C" void kernel_launch(void* const* d_in, const int* in_sizes, int n_in,
                              void* d_out, int out_size, void* d_ws, size_t ws_size,
                              hipStream_t stream) {
  const float* x    = (const float*)d_in[0];
  const float* cls  = (const float*)d_in[1];
  const float* Wk   = (const float*)d_in[2];
  const float* Wv   = (const float*)d_in[3];
  const float* Wlin = (const float*)d_in[4];
  const float* blin = (const float*)d_in[5];

  float* ws = (float*)d_ws;
  float*     psum   = ws + OFF_PSUM;
  float*     psumsq = ws + OFF_PSUMSQ;
  float*     mean   = ws + OFF_MEAN;
  float*     rstd   = ws + OFF_RSTD;
  float*     bias   = ws + OFF_BIAS;
  _Float16*  W2h    = (_Float16*)(ws + OFF_W2H);
  _Float16*  W2f    = (_Float16*)(ws + OFF_W2F);
  float*     attn   = ws + OFF_ATTN;
  float*     pooled = ws + OFF_POOL;

  float* out  = (float*)d_out;                       // [32, 1000]
  float* xcls = (float*)d_out + (long)BB * OUTD;     // [32, 768]

  // 1) BN partial stats (deterministic two-stage reduction)
  k_bn_partial<<<STATS_BLOCKS, 256, 0, stream>>>(x, psum, psumsq);

  // 2) finalize mean/rstd, fold Wk*q*scale*rstd into f16 W2
  k_finalize<<<3, 256, 0, stream>>>(Wk, cls, psum, psumsq, mean, rstd, W2h);

  // 3) bias + B-fragment swizzle
  k_bias<<<1, HP, 0, stream>>>(mean, W2h, bias);
  k_swizzle_w2<<<3, 256, 0, stream>>>(W2h, W2f);

  // 4) scores via WMMA: 8192 waves -> 2048 blocks x 128 threads
  k_scores_wmma<<<2048, 128, 0, stream>>>(x, W2f, bias, attn);

  // 5) softmax per (b, padded head): 512 blocks
  k_softmax<<<BB * HP, 256, 0, stream>>>(attn);

  // 6) pooled via WMMA: 32*48 = 1536 waves -> 384 blocks x 128 threads
  k_pool_wmma<<<384, 128, 0, stream>>>(x, attn, mean, rstd, pooled);

  // 7) x_cls: 32*768 outputs
  k_xcls<<<(BB * CC + 255) / 256, 256, 0, stream>>>(pooled, Wv, xcls);

  // 8) classifier: 32*1000 outputs
  k_out<<<(BB * OUTD + 255) / 256, 256, 0, stream>>>(xcls, Wlin, blin, out);
}
